// SelfAttentionQuantQPyTorch_54271206752593
// MI455X (gfx1250) — matrix operand order repaired and checked by reading, not compiled, not run on hardware
//
#include <hip/hip_runtime.h>
#include <hip/hip_bf16.h>

// ---------------------------------------------------------------------------
// MI455X (gfx1250, wave32) implementation of quantized self-attention.
// Heavy math: v_wmma_f32_16x16x32_f16. GEMM tiles staged via CDNA5 async
// global->LDS copies (global_load_async_to_lds_b128, ASYNCcnt) with explicit
// LDS double-buffering.
// ---------------------------------------------------------------------------

typedef __attribute__((ext_vector_type(16))) _Float16 v16h;
typedef __attribute__((ext_vector_type(8)))  _Float16 v8h;
typedef __attribute__((ext_vector_type(4)))  _Float16 v4h;
typedef __attribute__((ext_vector_type(8)))  float    v8f;

#define C_DIM   512
#define N_BATCH 4096
#define S_SEQ   31
#define H_HEADS 8
#define D_HEAD  64
#define M_ROWS  (N_BATCH * S_SEQ)   // 126976
#define EPS_BN  1e-5f

__device__ __forceinline__ float quantf(float x) {
  // round(x * 2^40) / 2^40, clip to [-2^23, 2^23 - 2^-40]
  float r = rintf(x * 1.0995116277760000e12f) * 9.0949470177292824e-13f;
  return fminf(fmaxf(r, -8388608.0f), 8388608.0f);
}

// ---- fragment loaders (16-bit A 16x32 and B 32x16 per CDNA5 ISA layouts) ----
// A: lane l(<16)=row l holds K 0..7 & 16..23 ; lane 16+l holds K 8..15 & 24..31
__device__ __forceinline__ v16h loadFragA(const _Float16* p) {
  union { v16h v; v8h h[2]; } u;
  u.h[0] = *(const v8h*)(p);        // K k0..k0+7
  u.h[1] = *(const v8h*)(p + 16);   // K k0+16..k0+23
  return u.v;
}
// B: lane l holds column l%16, K 0..7 & 16..23 (l<16) ; K 8..15 & 24..31 (l>=16)
__device__ __forceinline__ v16h loadFragB(const _Float16* p) {
  union { v16h v; v8h h[2]; } u;
  u.h[0] = *(const v8h*)(p);
  u.h[1] = *(const v8h*)(p + 16);
  return u.v;
}

// ---- CDNA5 async global->LDS tile copy (tracked by ASYNCcnt) ---------------
// Per thread: 32B of the A tile (two b128, INST_OFFSET applies to both the
// global and LDS address, so chunk 2 is just offset:16) + 16B of the B tile.
// LDS byte offset = low 32 bits of the generic pointer (flat LDS aperture).
__device__ __forceinline__ void asyncTile(const _Float16* ga, _Float16* la,
                                          const _Float16* gb, _Float16* lb) {
  unsigned lao = (unsigned)(unsigned long long)la;
  unsigned lbo = (unsigned)(unsigned long long)lb;
  asm volatile("global_load_async_to_lds_b128 %0, %2, off\n\t"
               "global_load_async_to_lds_b128 %0, %2, off offset:16\n\t"
               "global_load_async_to_lds_b128 %1, %3, off"
               :: "v"(lao), "v"(lbo), "v"(ga), "v"(gb)
               : "memory");
}
__device__ __forceinline__ void waitAsync() {
  asm volatile("s_wait_asynccnt 0x0" ::: "memory");
}

// ---------------------------------------------------------------------------
// 1) BatchNorm(C) stats over (N,S): per-channel sum / sumsq via atomics
// ---------------------------------------------------------------------------
__global__ __launch_bounds__(256) void bn_stats_kernel(
    const float* __restrict__ x, float* __restrict__ sum, float* __restrict__ sq) {
  const int c = threadIdx.x;                 // channels c and c+256
  const long row0 = (long)blockIdx.x * 128;
  float s0 = 0.f, s1 = 0.f, q0 = 0.f, q1 = 0.f;
  for (int r = 0; r < 128; ++r) {
    const float* p = x + (row0 + r) * C_DIM;
    float a = p[c], b = p[c + 256];
    s0 += a; q0 += a * a;
    s1 += b; q1 += b * b;
  }
  atomicAdd(&sum[c], s0);        atomicAdd(&sq[c], q0);
  atomicAdd(&sum[c + 256], s1);  atomicAdd(&sq[c + 256], q1);
}

__global__ __launch_bounds__(256) void bn_finalize_kernel(
    const float* __restrict__ sum, const float* __restrict__ sq,
    const float* __restrict__ gamma, const float* __restrict__ beta,
    float* __restrict__ scale, float* __restrict__ shift) {
  int c = blockIdx.x * 256 + threadIdx.x;
  if (c < C_DIM) {
    const float inv_n = 1.0f / (float)M_ROWS;
    float mu  = sum[c] * inv_n;
    float var = fmaxf(sq[c] * inv_n - mu * mu, 0.0f);
    float is  = rsqrtf(var + EPS_BN);
    float sc  = is * gamma[c];
    scale[c] = sc;
    shift[c] = beta[c] - mu * sc;
  }
}

// ---------------------------------------------------------------------------
// 2) normalize + quant + cast to f16   (xn = quant((x-mu)*inv*g + b))
// ---------------------------------------------------------------------------
__global__ __launch_bounds__(256) void normalize_kernel(
    const float* __restrict__ x, const float* __restrict__ scale,
    const float* __restrict__ shift, _Float16* __restrict__ xn) {
  size_t i = ((size_t)blockIdx.x * 256 + threadIdx.x) * 4;
  int c = (int)(i & (C_DIM - 1));
  float4 v = *(const float4*)(x + i);
  v4h o;
  o[0] = (_Float16)quantf(v.x * scale[c + 0] + shift[c + 0]);
  o[1] = (_Float16)quantf(v.y * scale[c + 1] + shift[c + 1]);
  o[2] = (_Float16)quantf(v.z * scale[c + 2] + shift[c + 2]);
  o[3] = (_Float16)quantf(v.w * scale[c + 3] + shift[c + 3]);
  *(v4h*)(xn + i) = o;
}

__global__ __launch_bounds__(256) void cast_f32_f16_kernel(
    const float* __restrict__ src, _Float16* __restrict__ dst, int n) {
  int i = blockIdx.x * 256 + threadIdx.x;
  if (i < n) dst[i] = (_Float16)src[i];
}

// ---------------------------------------------------------------------------
// 3) WMMA GEMM: out[m,j] = quant(sum_c A[m,c] * W[j,c] (+bias) (+resid))
//    Block tile 128x64, 8 waves (4x2), wave tile 32x32, K-step 32.
//    Tiles staged with async global->LDS copies, LDS double-buffered:
//    one barrier + one s_wait_asynccnt per K-step.
//    mode 0: f16 [M,512] row-major (Q,K)
//    mode 1: f16 Vt layout [n][h][d][l-pad32] (V)
//    mode 2: f32 out = quant(quant(acc + bias) + resid)   (final projection)
// ---------------------------------------------------------------------------
__global__ __launch_bounds__(256) void gemm_kernel(
    const _Float16* __restrict__ A,   // [M, 512] f16
    const _Float16* __restrict__ Bw,  // [512, 512] f16, row j contiguous in c
    _Float16* __restrict__ outH,
    const float* __restrict__ bias,
    const float* __restrict__ resid,
    float* __restrict__ outF,
    int mode) {
  __shared__ _Float16 As[2][128 * 40];  // stride 40 halves (80B, 16B-aligned)
  __shared__ _Float16 Bs[2][64 * 40];

  const int tid   = threadIdx.x;
  const long mBase = (long)blockIdx.x * 128;
  const int  nBase = blockIdx.y * 64;

  const int w  = tid >> 5;
  const int ln = tid & 31;
  const int wm = w >> 1, wn = w & 1;      // 4 x 2 waves
  const int m0 = wm * 32, n0 = wn * 32;
  const int ll = ln & 15, lg = ln >> 4;

  // cooperative async tile loaders
  const int arow = tid >> 1, aseg = tid & 1;   // 128 rows x 2 x 16 halves
  const int brow = tid >> 2, bseg = tid & 3;   // 64 rows x 4 x 8 halves
  const _Float16* aSrc = A  + (mBase + arow) * C_DIM + aseg * 16;
  const _Float16* bSrc = Bw + (long)(nBase + brow) * C_DIM + bseg * 8;
  const int aOff = arow * 40 + aseg * 16;
  const int bOff = brow * 40 + bseg * 8;

  v8f acc[2][2] = {};

  // prologue: stage tile 0 into buffer 0
  asyncTile(aSrc, &As[0][aOff], bSrc, &Bs[0][bOff]);
  waitAsync();
  __syncthreads();

  for (int kt = 0; kt < C_DIM; kt += 32) {
    const int cur = (kt >> 5) & 1;
    const int nxt = cur ^ 1;

    if (kt + 32 < C_DIM) {
      asyncTile(aSrc + kt + 32, &As[nxt][aOff], bSrc + kt + 32, &Bs[nxt][bOff]);
      if (kt + 64 < C_DIM) {
        __builtin_prefetch(aSrc + kt + 64, 0, 1);   // global_prefetch_b8 (L2)
        __builtin_prefetch(bSrc + kt + 64, 0, 1);
      }
    }

    const _Float16* AsC = As[cur];
    const _Float16* BsC = Bs[cur];
    v16h a0 = loadFragA(AsC + (m0 + ll) * 40 + lg * 8);
    v16h a1 = loadFragA(AsC + (m0 + 16 + ll) * 40 + lg * 8);
    v16h b0 = loadFragB(BsC + (n0 + ll) * 40 + lg * 8);
    v16h b1 = loadFragB(BsC + (n0 + 16 + ll) * 40 + lg * 8);

    acc[0][0] = __builtin_amdgcn_wmma_f32_16x16x32_f16(false, a0, false, b0, (short)0, acc[0][0], false, false);
    acc[0][1] = __builtin_amdgcn_wmma_f32_16x16x32_f16(false, a0, false, b1, (short)0, acc[0][1], false, false);
    acc[1][0] = __builtin_amdgcn_wmma_f32_16x16x32_f16(false, a1, false, b0, (short)0, acc[1][0], false, false);
    acc[1][1] = __builtin_amdgcn_wmma_f32_16x16x32_f16(false, a1, false, b1, (short)0, acc[1][1], false, false);

    waitAsync();        // our async writes into buf[nxt] are done
    __syncthreads();    // everyone's writes done; everyone's reads of cur done
  }

  // epilogue — C/D layout: VGPR r, lane: row = r + 8*lg, col = ll
  #pragma unroll
  for (int mt = 0; mt < 2; ++mt) {
    #pragma unroll
    for (int nt = 0; nt < 2; ++nt) {
      const int col = nBase + n0 + nt * 16 + ll;
      #pragma unroll
      for (int r = 0; r < 8; ++r) {
        const long row = mBase + m0 + mt * 16 + lg * 8 + r;
        const float val = acc[mt][nt][r];
        if (mode == 2) {
          float t = quantf(val + bias[col]);
          float o = quantf(t + resid[row * C_DIM + col]);
          outF[row * C_DIM + col] = o;
        } else if (mode == 1) {
          int n = (int)(row / S_SEQ), l = (int)(row % S_SEQ);
          int h = col >> 6, d = col & 63;
          outH[(((long)(n * H_HEADS + h) * D_HEAD + d) << 5) + l] = (_Float16)quantf(val);
        } else {
          outH[row * C_DIM + col] = (_Float16)quantf(val);
        }
      }
    }
  }
}

// ---------------------------------------------------------------------------
// 4) energy[n,h,q,k] = quant(sum_d Q[n,q,h,d] * K[n,k,h,d])  (31x31, pad 32)
//    one wave per (n,h); 2x2 tiles x 2 k-steps = 8 WMMAs
// ---------------------------------------------------------------------------
__global__ __launch_bounds__(256) void energy_kernel(
    const _Float16* __restrict__ Q, const _Float16* __restrict__ K,
    float* __restrict__ E) {
  const int n  = blockIdx.x;
  const int h  = threadIdx.x >> 5;
  const int ln = threadIdx.x & 31, ll = ln & 15, lg = ln >> 4;
  const v16h z = {};

  v8f acc[2][2] = {};
  #pragma unroll
  for (int ks = 0; ks < 2; ++ks) {
    v16h a[2], b[2];
    #pragma unroll
    for (int mt = 0; mt < 2; ++mt) {
      int q = mt * 16 + ll;
      if (q < S_SEQ) {
        const _Float16* p = Q + ((long)n * S_SEQ + q) * C_DIM + h * D_HEAD + ks * 32 + lg * 8;
        a[mt] = loadFragA(p);
      } else a[mt] = z;
    }
    #pragma unroll
    for (int nt = 0; nt < 2; ++nt) {
      int kc = nt * 16 + ll;
      if (kc < S_SEQ) {
        const _Float16* p = K + ((long)n * S_SEQ + kc) * C_DIM + h * D_HEAD + ks * 32 + lg * 8;
        b[nt] = loadFragB(p);
      } else b[nt] = z;
    }
    #pragma unroll
    for (int mt = 0; mt < 2; ++mt)
      #pragma unroll
      for (int nt = 0; nt < 2; ++nt)
        acc[mt][nt] = __builtin_amdgcn_wmma_f32_16x16x32_f16(false, a[mt], false, b[nt], (short)0, acc[mt][nt], false, false);
  }

  const long base = ((long)n * H_HEADS + h) * S_SEQ * S_SEQ;
  #pragma unroll
  for (int mt = 0; mt < 2; ++mt)
    #pragma unroll
    for (int nt = 0; nt < 2; ++nt) {
      int kc = nt * 16 + ll;
      #pragma unroll
      for (int r = 0; r < 8; ++r) {
        int q = mt * 16 + lg * 8 + r;
        if (q < S_SEQ && kc < S_SEQ)
          E[base + (long)q * S_SEQ + kc] = quantf(acc[mt][nt][r]);
      }
    }
}

// ---------------------------------------------------------------------------
// 5) BN over energy: channels = S*S = 961, stats over (N*H) = 32768 rows
// ---------------------------------------------------------------------------
__global__ __launch_bounds__(256) void ebn_stats_kernel(
    const float* __restrict__ E, float* __restrict__ sum, float* __restrict__ sq) {
  const int t = threadIdx.x;
  const long r0 = (long)blockIdx.x * 64;
  float s[4] = {0, 0, 0, 0}, q[4] = {0, 0, 0, 0};
  for (int r = 0; r < 64; ++r) {
    const float* p = E + (r0 + r) * (S_SEQ * S_SEQ);
    #pragma unroll
    for (int j = 0; j < 4; ++j) {
      int ch = t + j * 256;
      if (ch < S_SEQ * S_SEQ) { float v = p[ch]; s[j] += v; q[j] += v * v; }
    }
  }
  #pragma unroll
  for (int j = 0; j < 4; ++j) {
    int ch = t + j * 256;
    if (ch < S_SEQ * S_SEQ) { atomicAdd(&sum[ch], s[j]); atomicAdd(&sq[ch], q[j]); }
  }
}

__global__ __launch_bounds__(256) void ebn_finalize_kernel(
    const float* __restrict__ sum, const float* __restrict__ sq,
    const float* __restrict__ gamma, const float* __restrict__ beta,
    float* __restrict__ scale, float* __restrict__ shift) {
  int c = blockIdx.x * 256 + threadIdx.x;
  if (c < S_SEQ * S_SEQ) {
    const float inv_n = 1.0f / (float)(N_BATCH * H_HEADS);
    float mu  = sum[c] * inv_n;
    float var = fmaxf(sq[c] * inv_n - mu * mu, 0.0f);
    float is  = rsqrtf(var + EPS_BN);
    float sc  = is * gamma[c];
    scale[c] = sc;
    shift[c] = beta[c] - mu * sc;
  }
}

// ---------------------------------------------------------------------------
// 6) fused energy-BN + softmax(/sqrt(C)) + quant -> attn f16 [N,H,32,32]
//    one block per (n,h); wave handles rows q = w, w+8, w+16, w+24
// ---------------------------------------------------------------------------
__global__ __launch_bounds__(256) void softmax_kernel(
    const float* __restrict__ E, const float* __restrict__ scale,
    const float* __restrict__ shift, _Float16* __restrict__ attn) {
  const long nh = blockIdx.x;
  const int  w  = threadIdx.x >> 5;
  const int  k  = threadIdx.x & 31;
  const float invSqrtC = 0.044194173824159216f;  // 1/sqrt(512)

  #pragma unroll
  for (int i = 0; i < 4; ++i) {
    const int q = w + i * 8;                 // 0..31 (31 = pad row)
    _Float16 outv = (_Float16)0.0f;
    if (q < S_SEQ) {
      const bool valid = k < S_SEQ;
      float v = -3.0e38f;
      if (valid) {
        int ch = q * S_SEQ + k;
        float e = E[(nh * S_SEQ + q) * S_SEQ + k];
        float en = quantf(e * scale[ch] + shift[ch]);
        v = en * invSqrtC;
      }
      float m = v;
      #pragma unroll
      for (int o = 16; o > 0; o >>= 1) m = fmaxf(m, __shfl_xor(m, o, 32));
      float ex = valid ? __expf(v - m) : 0.0f;
      float s = ex;
      #pragma unroll
      for (int o = 16; o > 0; o >>= 1) s += __shfl_xor(s, o, 32);
      if (valid) outv = (_Float16)quantf(ex / s);
    }
    attn[(nh * 32 + q) * 32 + k] = outv;     // pad col/row written as 0
  }
}

// ---------------------------------------------------------------------------
// 7) ctx[n,q,h,d] = quant(sum_l attn[n,h,q,l] * V[n,l,h,d])
//    one wave per (n,h); single K-step (l padded to 32), 2x4 tiles = 8 WMMAs
//    B operand read contiguously from Vt[n][h][d][l]
// ---------------------------------------------------------------------------
__global__ __launch_bounds__(256) void attnv_kernel(
    const _Float16* __restrict__ attn, const _Float16* __restrict__ Vt,
    _Float16* __restrict__ Ctx) {
  const int n  = blockIdx.x;
  const int h  = threadIdx.x >> 5;
  const int ln = threadIdx.x & 31, ll = ln & 15, lg = ln >> 4;
  const long nh = (long)n * H_HEADS + h;

  v16h a[2];
  #pragma unroll
  for (int mt = 0; mt < 2; ++mt)
    a[mt] = loadFragA(attn + (nh * 32 + mt * 16 + ll) * 32 + lg * 8);

  v8f acc[2][4] = {};
  #pragma unroll
  for (int nt = 0; nt < 4; ++nt) {
    v16h b = loadFragB(Vt + (nh * D_HEAD + nt * 16 + ll) * 32 + lg * 8);
    acc[0][nt] = __builtin_amdgcn_wmma_f32_16x16x32_f16(false, a[0], false, b, (short)0, acc[0][nt], false, false);
    acc[1][nt] = __builtin_amdgcn_wmma_f32_16x16x32_f16(false, a[1], false, b, (short)0, acc[1][nt], false, false);
  }

  #pragma unroll
  for (int mt = 0; mt < 2; ++mt)
    #pragma unroll
    for (int nt = 0; nt < 4; ++nt) {
      int d = nt * 16 + ll;
      #pragma unroll
      for (int r = 0; r < 8; ++r) {
        int q = mt * 16 + lg * 8 + r;
        if (q < S_SEQ)
          Ctx[((long)n * S_SEQ + q) * C_DIM + h * D_HEAD + d] = (_Float16)quantf(acc[mt][nt][r]);
      }
    }
}

// ---------------------------------------------------------------------------
extern "C" void kernel_launch(void* const* d_in, const int* in_sizes, int n_in,
                              void* d_out, int out_size, void* d_ws, size_t ws_size,
                              hipStream_t stream) {
  const float* x     = (const float*)d_in[0];
  const float* gamma = (const float*)d_in[1];
  const float* beta  = (const float*)d_in[2];
  const float* wq    = (const float*)d_in[3];
  const float* wk    = (const float*)d_in[4];
  const float* wv    = (const float*)d_in[5];
  const float* wo    = (const float*)d_in[6];
  const float* bo    = (const float*)d_in[7];
  const float* pg    = (const float*)d_in[8];
  const float* pb    = (const float*)d_in[9];
  float* out = (float*)d_out;

  char* ws = (char*)d_ws;
  float* c_sum   = (float*)(ws);           // 512 f32
  float* c_sq    = (float*)(ws + 2048);
  float* c_scale = (float*)(ws + 4096);
  float* c_shift = (float*)(ws + 6144);
  float* e_sum   = (float*)(ws + 8192);    // 1024 f32 (961 used)
  float* e_sq    = (float*)(ws + 12288);
  float* e_scale = (float*)(ws + 16384);
  float* e_shift = (float*)(ws + 20480);

  size_t off = 32768;
  const size_t szMxC  = (size_t)M_ROWS * C_DIM * 2;                       // f16 [M,512]
  const size_t szW    = (size_t)C_DIM * C_DIM * 2;                        // f16 [512,512]
  const size_t szVt   = (size_t)N_BATCH * H_HEADS * D_HEAD * 32 * 2;      // f16 [n,h,d,32]
  const size_t szE    = (size_t)N_BATCH * H_HEADS * S_SEQ * S_SEQ * 4;    // f32
  const size_t szAttn = (size_t)N_BATCH * H_HEADS * 32 * 32 * 2;          // f16

  _Float16* Xn  = (_Float16*)(ws + off); off += szMxC;
  _Float16* Wqh = (_Float16*)(ws + off); off += szW;
  _Float16* Wkh = (_Float16*)(ws + off); off += szW;
  _Float16* Wvh = (_Float16*)(ws + off); off += szW;
  _Float16* Woh = (_Float16*)(ws + off); off += szW;
  _Float16* Qh  = (_Float16*)(ws + off); off += szMxC;
  _Float16* Kh  = (_Float16*)(ws + off); off += szMxC;
  _Float16* Vt  = (_Float16*)(ws + off); off += szVt;
  float*    E   = (float*)   (ws + off); off += szE;
  _Float16* At  = (_Float16*)(ws + off); off += szAttn;
  _Float16* Ctx = (_Float16*)(ws + off); off += szMxC;

  // zero atomic accumulators + Vt pad (pad column l=31 must be 0-safe)
  hipMemsetAsync(c_sum, 0, 4096, stream);
  hipMemsetAsync(e_sum, 0, 8192, stream);
  hipMemsetAsync(Vt, 0, szVt, stream);

  // 1) BN(C) stats + finalize
  bn_stats_kernel<<<M_ROWS / 128, 256, 0, stream>>>(x, c_sum, c_sq);
  bn_finalize_kernel<<<2, 256, 0, stream>>>(c_sum, c_sq, gamma, beta, c_scale, c_shift);

  // 2) weight casts + normalize/cast input
  cast_f32_f16_kernel<<<1024, 256, 0, stream>>>(wq, Wqh, C_DIM * C_DIM);
  cast_f32_f16_kernel<<<1024, 256, 0, stream>>>(wk, Wkh, C_DIM * C_DIM);
  cast_f32_f16_kernel<<<1024, 256, 0, stream>>>(wv, Wvh, C_DIM * C_DIM);
  cast_f32_f16_kernel<<<1024, 256, 0, stream>>>(wo, Woh, C_DIM * C_DIM);
  normalize_kernel<<<(M_ROWS * C_DIM / 4) / 256, 256, 0, stream>>>(x, c_scale, c_shift, Xn);

  // 3) Q,K,V projections (WMMA GEMM 126976x512x512, async-LDS staged)
  dim3 gg(M_ROWS / 128, C_DIM / 64);
  gemm_kernel<<<gg, 256, 0, stream>>>(Xn, Wqh, Qh, nullptr, nullptr, nullptr, 0);
  gemm_kernel<<<gg, 256, 0, stream>>>(Xn, Wkh, Kh, nullptr, nullptr, nullptr, 0);
  gemm_kernel<<<gg, 256, 0, stream>>>(Xn, Wvh, Vt, nullptr, nullptr, nullptr, 1);

  // 4) energy = Q K^T per (n,h)
  energy_kernel<<<N_BATCH, 256, 0, stream>>>(Qh, Kh, E);

  // 5) BN over energy (channels = 961)
  ebn_stats_kernel<<<(N_BATCH * H_HEADS) / 64, 256, 0, stream>>>(E, e_sum, e_sq);
  ebn_finalize_kernel<<<4, 256, 0, stream>>>(e_sum, e_sq, pg, pb, e_scale, e_shift);

  // 6) fused BN + softmax + quant
  softmax_kernel<<<N_BATCH * H_HEADS, 256, 0, stream>>>(E, e_scale, e_shift, At);

  // 7) ctx = attn @ V
  attnv_kernel<<<N_BATCH, 256, 0, stream>>>(At, Vt, Ctx);

  // 8) out = quant(quant(ctx @ wo^T + bo) + x)
  gemm_kernel<<<gg, 256, 0, stream>>>(Ctx, Woh, nullptr, bo, x, out, 2);

  (void)in_sizes; (void)n_in; (void)out_size; (void)ws_size;
}
